// SmartVocoder_90366111908356
// MI455X (gfx1250) — matrix operand order repaired and checked by reading, not compiled, not run on hardware
//
#include <hip/hip_runtime.h>
#include <hip/hip_bf16.h>
#include <math.h>

// ---------------------------------------------------------------------------
// SmartVocoder (WaveGlow-style flow) forward for gfx1250.
// Heavy convs (dilated 128->256, res/skip 128->256, cond 80->1536 slice) are
// WMMA f16 GEMMs with f32 accumulation (v_wmma_f32_16x16x32_f16), blocked
// 16M x 64N per wave for A-fragment reuse (4 WMMAs per A load).
// ---------------------------------------------------------------------------

typedef __attribute__((ext_vector_type(16))) _Float16 v16h;
typedef __attribute__((ext_vector_type(8)))  float    v8f;

#define WMMA16(a, b, c) \
    __builtin_amdgcn_wmma_f32_16x16x32_f16(false, (a), false, (b), (short)0, (c), false, false)

// ---------------- upsample: 1-channel transposed conv2d + leaky_relu(0.4) ----
__device__ __forceinline__ int fdiv_floor(int a, int b) {
    int q = a / b;
    if ((a % b) && ((a ^ b) < 0)) q--;
    return q;
}

__global__ void k_upsample(const float* __restrict__ in, const float* __restrict__ w,
                           const float* __restrict__ bias, float* __restrict__ out,
                           int B, int H, int Win, int u, int pw, int Tout) {
    long idx = (long)blockIdx.x * blockDim.x + threadIdx.x;
    long total = (long)B * H * Tout;
    if (idx >= total) return;
    int t = (int)(idx % Tout);
    int h = (int)((idx / Tout) % H);
    int b = (int)(idx / ((long)Tout * H));
    int ku = 2 * u;
    // need kw = j*u - t + pw in [0, 2u)
    int jlo = fdiv_floor(t - pw + u - 1, u);     // ceil((t-pw)/u)
    int jhi = fdiv_floor(t - pw + ku - 1, u);
    if (jlo < 0) jlo = 0;
    if (jhi > Win - 1) jhi = Win - 1;
    float acc = 0.f;
    for (int kh = 0; kh < 3; ++kh) {
        int hi = h + kh - 1;
        if (hi < 0 || hi >= H) continue;
        const float* row = in + ((size_t)b * H + hi) * Win;
        const float* wr  = w + (2 - kh) * ku;
        for (int j = jlo; j <= jhi; ++j) {
            int kw = j * u - t + pw;
            if (kw >= 0 && kw < ku) acc += row[j] * wr[ku - 1 - kw];
        }
    }
    acc += bias[0];
    out[idx] = (acc > 0.f) ? acc : 0.4f * acc;
}

// ---------------- squeeze_ch for input x: (2,1,40960) -> (2,8,5120) ----------
__global__ void k_squeeze_in(const float* __restrict__ x, float* __restrict__ out) {
    int idx = blockIdx.x * blockDim.x + threadIdx.x;
    if (idx >= 2 * 8 * 5120) return;
    int t = idx % 5120;
    int s = (idx / 5120) % 8;
    int b = idx / (8 * 5120);
    out[idx] = x[(size_t)b * 40960 + t * 8 + s];
}

// ---------------- squeeze_batch: (B,8,T) -> (2B,8,T/2) -----------------------
__global__ void k_squeeze_batch(const float* __restrict__ in, float* __restrict__ out,
                                int B, int T) {
    int idx = blockIdx.x * blockDim.x + threadIdx.x;
    int Th = T >> 1;
    int total = 2 * B * 8 * Th;
    if (idx >= total) return;
    int t  = idx % Th;
    int c  = (idx / Th) % 8;
    int nb = idx / (8 * Th);
    int b = nb >> 1, s = nb & 1;
    out[idx] = in[((size_t)b * 8 + c) * T + 2 * t + s];
}

// ---------------- actnorm + invertible 1x1 conv (8x8), in place -------------
__global__ void k_actnorm_inv(float* __restrict__ io, const float* __restrict__ loc,
                              const float* __restrict__ scale, const float* __restrict__ invw,
                              int B, int T) {
    long idx = (long)blockIdx.x * blockDim.x + threadIdx.x;
    if (idx >= (long)B * T) return;
    int t = (int)(idx % T);
    int b = (int)(idx / T);
    float v[8];
    for (int i = 0; i < 8; ++i)
        v[i] = scale[i] * (io[((size_t)b * 8 + i) * T + t] + loc[i]);
    for (int o = 0; o < 8; ++o) {
        float a = 0.f;
        for (int i = 0; i < 8; ++i) a += invw[o * 8 + i] * v[i];
        io[((size_t)b * 8 + o) * T + t] = a;
    }
}

// ---------------- per-flow logdet: actnorm sum + 8x8 slogdet (1 thread) -----
__global__ void k_flow_logdet(const float* __restrict__ scale, const float* __restrict__ invw,
                              float* __restrict__ ldet, float BT) {
    float s = 0.f;
    for (int i = 0; i < 8; ++i) s += logf(fabsf(scale[i]));
    float Mx[64];
    for (int i = 0; i < 64; ++i) Mx[i] = invw[i];
    float ld = 0.f;
    for (int c = 0; c < 8; ++c) {
        int p = c; float mx = fabsf(Mx[c * 8 + c]);
        for (int r = c + 1; r < 8; ++r) {
            float a = fabsf(Mx[r * 8 + c]);
            if (a > mx) { mx = a; p = r; }
        }
        if (p != c)
            for (int cc = 0; cc < 8; ++cc) {
                float tmp = Mx[c * 8 + cc]; Mx[c * 8 + cc] = Mx[p * 8 + cc]; Mx[p * 8 + cc] = tmp;
            }
        float piv = Mx[c * 8 + c];
        ld += logf(fabsf(piv));
        for (int r = c + 1; r < 8; ++r) {
            float f = Mx[r * 8 + c] / piv;
            for (int cc = c; cc < 8; ++cc) Mx[r * 8 + cc] -= f * Mx[c * 8 + cc];
        }
    }
    *ldet += BT * (s + ld);
}

// ---------------- start conv: 4 -> 128 (1x1) --------------------------------
__global__ void k_start(const float* __restrict__ outbuf, const float* __restrict__ sw,
                        const float* __restrict__ sb, float* __restrict__ audio,
                        int B, int T) {
    long idx = (long)blockIdx.x * blockDim.x + threadIdx.x;
    if (idx >= (long)B * 128 * T) return;
    int t = (int)(idx % T);
    int o = (int)((idx / T) % 128);
    int b = (int)(idx / ((long)128 * T));
    float acc = sb[o];
    for (int i = 0; i < 4; ++i)
        acc += sw[o * 4 + i] * outbuf[((size_t)b * 8 + i) * T + t];
    audio[((size_t)b * 128 + o) * T + t] = acc;
}

// ---------------- pack weight (M,K) f32 -> WMMA A fragments (16x32 f16) -----
// A layout: lane<16: m=mt*16+lane,    elem i -> k = (i<8? i   : i+8)
//           lane>=16: m=mt*16+lane-16, elem i -> k = (i<8? i+8 : i+16)
__global__ void k_packA(const float* __restrict__ w, _Float16* __restrict__ frag,
                        int Mtiles, int Kt, int rowStride, int colStride, int Kreal) {
    int tid = blockIdx.x * blockDim.x + threadIdx.x;
    int total = Mtiles * Kt * 32;
    if (tid >= total) return;
    int lane = tid & 31;
    int f = tid >> 5;                  // f = mt*Kt + kt
    int kt = f % Kt;
    int mt = f / Kt;
    int m = (mt << 4) + (lane & 15);
    int hi = (lane < 16) ? 0 : 1;
    _Float16* dst = frag + ((size_t)f * 32 + lane) * 16;
    for (int i = 0; i < 16; ++i) {
        int kk = hi ? ((i < 8) ? i + 8 : i + 16) : ((i < 8) ? i : i + 8);
        int k = (kt << 5) + kk;
        float v = (k < Kreal) ? w[(size_t)m * rowStride + (size_t)k * colStride] : 0.f;
        dst[i] = (_Float16)v;
    }
}

// ---------------- pack activations (B,K,T) f32 -> WMMA B fragments (32x16) --
// B layout: n = nt*16 + lane%16; elem i -> k = kt*32 + (lane<16 ? 0 : 8) + (i<8 ? i : i+8)
// with time shift + zero padding (for dilated conv taps).
__global__ void k_packB(const float* __restrict__ act, _Float16* __restrict__ frag,
                        int Bn, int Kt, int Kreal, int T, int shift) {
    int tid = blockIdx.x * blockDim.x + threadIdx.x;
    int ntiles = T >> 4;
    int total = Bn * ntiles * Kt * 32;
    if (tid >= total) return;
    int lane = tid & 31;
    int f = tid >> 5;                  // f = (b*ntiles + nt)*Kt + kt
    int kt = f % Kt;
    int nt = (f / Kt) % ntiles;
    int b  = f / (Kt * ntiles);
    int n = (nt << 4) + (lane & 15);
    int kb = (kt << 5) + ((lane < 16) ? 0 : 8);
    int t = n + shift;
    bool tin = (t >= 0 && t < T);
    _Float16* dst = frag + ((size_t)f * 32 + lane) * 16;
    for (int i = 0; i < 16; ++i) {
        int k = kb + ((i < 8) ? i : i + 8);
        float v = (tin && k < Kreal) ? act[((size_t)b * Kreal + k) * T + t] : 0.f;
        dst[i] = (_Float16)v;
    }
}

// ---------------- in_act GEMM: 3-tap dilated conv + cond GEMM, fused --------
// Wave computes a 16M x 64N block (4 N-tiles) so each A fragment is reused by
// 4 WMMAs. in_act = sum_r W_r@audio(shift_r) + Wc_slice@cond + in_b+cond_b+pos
__global__ void k_gemm_inact(const _Float16* __restrict__ Af, const _Float16* __restrict__ Bf,
                             size_t bShiftStride,
                             const _Float16* __restrict__ Afc, const _Float16* __restrict__ Bfc,
                             const float* __restrict__ inb, const float* __restrict__ condb,
                             int condOff, const float* __restrict__ pos, int g,
                             float* __restrict__ inact, int Bn, int T) {
    int lane = threadIdx.x & 31;
    int wv = threadIdx.x >> 5;
    int ntiles = T >> 4;
    int ngroups = ntiles >> 2;                       // 4 N-tiles per wave
    long tile = (long)blockIdx.x * (blockDim.x >> 5) + wv;
    long total = (long)Bn * 16 * ngroups;
    if (tile >= total) return;                       // whole-wave uniform exit
    int ng = (int)(tile % ngroups);
    int mt = (int)((tile / ngroups) % 16);
    int b  = (int)(tile / ((long)ngroups * 16));
    int nt0 = ng << 2;
    v8f acc0 = {0.f,0.f,0.f,0.f,0.f,0.f,0.f,0.f};
    v8f acc1 = acc0, acc2 = acc0, acc3 = acc0;
#pragma unroll
    for (int kt = 0; kt < 4; ++kt) {
#pragma unroll
        for (int r = 0; r < 3; ++r) {
            v16h a = *(const v16h*)(Af + (size_t)r * 32768 +
                                    ((size_t)(mt * 4 + kt) * 32 + lane) * 16);
            const _Float16* bp = Bf + (size_t)r * bShiftStride +
                                 (((size_t)(b * ntiles + nt0) * 4 + kt) * 32 + lane) * 16;
            acc0 = WMMA16(a, *(const v16h*)(bp),          acc0);   // nt step = 4*512
            acc1 = WMMA16(a, *(const v16h*)(bp + 2048),   acc1);
            acc2 = WMMA16(a, *(const v16h*)(bp + 4096),   acc2);
            acc3 = WMMA16(a, *(const v16h*)(bp + 6144),   acc3);
        }
    }
    int bu = b / g;
#pragma unroll
    for (int kt = 0; kt < 3; ++kt) {
        v16h a = *(const v16h*)(Afc + ((size_t)(mt * 3 + kt) * 32 + lane) * 16);
        const _Float16* bp = Bfc + (((size_t)(bu * ntiles + nt0) * 3 + kt) * 32 + lane) * 16;
        acc0 = WMMA16(a, *(const v16h*)(bp),          acc0);       // nt step = 3*512
        acc1 = WMMA16(a, *(const v16h*)(bp + 1536),   acc1);
        acc2 = WMMA16(a, *(const v16h*)(bp + 3072),   acc2);
        acc3 = WMMA16(a, *(const v16h*)(bp + 4608),   acc3);
    }
    int mb = (mt << 4) + ((lane < 16) ? 0 : 8);
    float badd[8];
#pragma unroll
    for (int j = 0; j < 8; ++j) {
        int m = mb + j;
        badd[j] = inb[m] + condb[condOff + m] +
                  (pos ? pos[(size_t)(b % g) * 1536 + condOff + m] : 0.f);
    }
    v8f* accs[4] = {&acc0, &acc1, &acc2, &acc3};
#pragma unroll
    for (int q = 0; q < 4; ++q) {
        int n = ((nt0 + q) << 4) + (lane & 15);
        v8f& ac = *accs[q];
#pragma unroll
        for (int j = 0; j < 8; ++j)
            inact[((size_t)b * 256 + mb + j) * T + n] = ac[j] + badd[j];
    }
}

// ---------------- gated activation ------------------------------------------
__global__ void k_gate(const float* __restrict__ inact, float* __restrict__ acts,
                       int Bn, int T) {
    long idx = (long)blockIdx.x * blockDim.x + threadIdx.x;
    if (idx >= (long)Bn * 128 * T) return;
    int t = (int)(idx % T);
    int m = (int)((idx / T) % 128);
    int b = (int)(idx / ((long)128 * T));
    float a1 = inact[((size_t)b * 256 + m) * T + t];
    float a2 = inact[((size_t)b * 256 + 128 + m) * T + t];
    acts[((size_t)b * 128 + m) * T + t] = tanhf(a1) * (1.f / (1.f + expf(-a2)));
}

// ---------------- res/skip GEMM with fused residual/skip update -------------
// 16M x 64N per wave; epilogue scatters rows to audio(residual)/skip buffers.
__global__ void k_gemm_rs(const _Float16* __restrict__ Af, const _Float16* __restrict__ Bf,
                          const float* __restrict__ rsb, float* __restrict__ audio,
                          float* __restrict__ skip, int Bn, int T, int Mtiles, int last) {
    int lane = threadIdx.x & 31;
    int wv = threadIdx.x >> 5;
    int ntiles = T >> 4;
    int ngroups = ntiles >> 2;
    long tile = (long)blockIdx.x * (blockDim.x >> 5) + wv;
    long total = (long)Bn * Mtiles * ngroups;
    if (tile >= total) return;
    int ng = (int)(tile % ngroups);
    int mt = (int)((tile / ngroups) % Mtiles);
    int b  = (int)(tile / ((long)ngroups * Mtiles));
    int nt0 = ng << 2;
    v8f acc0 = {0.f,0.f,0.f,0.f,0.f,0.f,0.f,0.f};
    v8f acc1 = acc0, acc2 = acc0, acc3 = acc0;
#pragma unroll
    for (int kt = 0; kt < 4; ++kt) {
        v16h a = *(const v16h*)(Af + ((size_t)(mt * 4 + kt) * 32 + lane) * 16);
        const _Float16* bp = Bf + (((size_t)(b * ntiles + nt0) * 4 + kt) * 32 + lane) * 16;
        acc0 = WMMA16(a, *(const v16h*)(bp),        acc0);
        acc1 = WMMA16(a, *(const v16h*)(bp + 2048), acc1);
        acc2 = WMMA16(a, *(const v16h*)(bp + 4096), acc2);
        acc3 = WMMA16(a, *(const v16h*)(bp + 6144), acc3);
    }
    int mb = (mt << 4) + ((lane < 16) ? 0 : 8);
    v8f* accs[4] = {&acc0, &acc1, &acc2, &acc3};
#pragma unroll
    for (int q = 0; q < 4; ++q) {
        int n = ((nt0 + q) << 4) + (lane & 15);
        v8f& ac = *accs[q];
#pragma unroll
        for (int j = 0; j < 8; ++j) {
            int m = mb + j;
            float v = ac[j] + rsb[m];
            if (last) {
                skip[((size_t)b * 128 + m) * T + n] += v;
            } else if (m < 128) {
                audio[((size_t)b * 128 + m) * T + n] += v;
            } else {
                skip[((size_t)b * 128 + (m - 128)) * T + n] += v;
            }
        }
    }
}

// ---------------- end conv (128->8) + affine coupling + log_s partial sums --
__global__ void k_end_coupling(const float* __restrict__ skip, const float* __restrict__ ew,
                               const float* __restrict__ eb, float* __restrict__ outbuf,
                               float* __restrict__ partials, int B, int T) {
    long idx = (long)blockIdx.x * blockDim.x + threadIdx.x;
    float lsum = 0.f;
    if (idx < (long)B * T) {
        int t = (int)(idx % T);
        int b = (int)(idx / T);
        float acc[8] = {0.f, 0.f, 0.f, 0.f, 0.f, 0.f, 0.f, 0.f};
        const float* sp = skip + (size_t)b * 128 * T + t;
        for (int k = 0; k < 128; ++k) {
            float s = sp[(size_t)k * T];
            for (int o = 0; o < 8; ++o) acc[o] += ew[o * 128 + k] * s;
        }
        for (int j = 0; j < 4; ++j) {
            float ls = acc[j] + eb[j];
            float bv = acc[4 + j] + eb[4 + j];
            size_t xi = ((size_t)b * 8 + 4 + j) * T + t;
            outbuf[xi] = expf(ls) * outbuf[xi] + bv;
            lsum += ls;
        }
    }
    __shared__ float sh[256];
    sh[threadIdx.x] = lsum;
    __syncthreads();
    for (int s = 128; s > 0; s >>= 1) {
        if ((int)threadIdx.x < s) sh[threadIdx.x] += sh[threadIdx.x + s];
        __syncthreads();
    }
    if (threadIdx.x == 0) partials[blockIdx.x] = sh[0];
}

__global__ void k_reduce(const float* __restrict__ partials, int n, float* __restrict__ ldet) {
    float s = 0.f;
    for (int i = 0; i < n; ++i) s += partials[i];
    *ldet += s;
}

// ---------------- finalize: out (81920) + logdet (1) ------------------------
__global__ void k_finalize(const float* __restrict__ cur, const float* __restrict__ ldet,
                           float* __restrict__ out) {
    int idx = blockIdx.x * blockDim.x + threadIdx.x;
    if (idx < 81920) out[idx] = cur[idx];
    else if (idx == 81920) out[idx] = *ldet;
}

// ===========================================================================

struct FlowP {
    const float *an_loc, *an_scale, *inv_w;
    const float *start_w, *start_b, *cond_w, *cond_b;
    const float *in_w[6], *in_b[6], *rs_w[6], *rs_b[6];
    const float *end_w, *end_b, *pos;
};

extern "C" void kernel_launch(void* const* d_in, const int* in_sizes, int n_in,
                              void* d_out, int out_size, void* d_ws, size_t ws_size,
                              hipStream_t stream) {
    (void)in_sizes; (void)n_in; (void)out_size; (void)ws_size;

    // -------- input walk (setup_inputs insertion order; None leaves skipped)
    const float* X = (const float*)d_in[0];      // (2,1,40960)
    const float* C = (const float*)d_in[1];      // (2,80,160)
    int p = 2;
    const float *ups_w[3], *ups_b[3];
    for (int i = 0; i < 3; ++i) {
        ups_w[i] = (const float*)d_in[p++];
        ups_b[i] = (const float*)d_in[p++];
    }
    FlowP fl[9];
    for (int blk = 0; blk < 3; ++blk)
        for (int f = 0; f < 3; ++f) {
            FlowP& F = fl[blk * 3 + f];
            F.an_loc   = (const float*)d_in[p++];
            F.an_scale = (const float*)d_in[p++];
            F.inv_w    = (const float*)d_in[p++];
            F.start_w  = (const float*)d_in[p++];
            F.start_b  = (const float*)d_in[p++];
            F.cond_w   = (const float*)d_in[p++];
            F.cond_b   = (const float*)d_in[p++];
            for (int l = 0; l < 6; ++l) F.in_w[l] = (const float*)d_in[p++];
            for (int l = 0; l < 6; ++l) F.in_b[l] = (const float*)d_in[p++];
            for (int l = 0; l < 6; ++l) F.rs_w[l] = (const float*)d_in[p++];
            for (int l = 0; l < 6; ++l) F.rs_b[l] = (const float*)d_in[p++];
            F.end_w = (const float*)d_in[p++];
            F.end_b = (const float*)d_in[p++];
            F.pos   = (blk > 0) ? (const float*)d_in[p++] : nullptr;
        }

    // -------- workspace carving
    char* base = (char*)d_ws;
    size_t off = 0;
    auto A = [&](size_t bytes) -> void* {
        void* r = base + off;
        off = (off + bytes + 255) & ~(size_t)255;
        return r;
    };
    int clT[3] = {1280, 2560, 5120};
    float* cl[3];
    for (int i = 0; i < 3; ++i) cl[i] = (float*)A((size_t)2 * 80 * clT[i] * 4);
    float* bufA  = (float*)A(81920 * 4);
    float* bufB  = (float*)A(81920 * 4);
    float* audio = (float*)A((size_t)1310720 * 4);         // (B,128,T), B*T=10240
    float* skip  = (float*)A((size_t)1310720 * 4);
    float* inact = (float*)A((size_t)2621440 * 4);         // (B,256,T)
    float* acts  = (float*)A((size_t)1310720 * 4);
    size_t bShiftStride = 1310720;                          // halves per shift buffer
    _Float16* BfAud  = (_Float16*)A((size_t)3 * bShiftStride * 2);
    _Float16* BfActs = (_Float16*)A((size_t)bShiftStride * 2);
    _Float16* BfCond = (_Float16*)A((size_t)2 * 96 * 5120 * 2); // (2,96,Tmax)
    _Float16* AfIn   = (_Float16*)A((size_t)3 * 32768 * 2);     // 3 taps, 16x4 frags
    _Float16* AfRs   = (_Float16*)A((size_t)32768 * 2);
    _Float16* AfCond = (_Float16*)A((size_t)16 * 3 * 512 * 2);  // 256x96 layer slice
    float* partials  = (float*)A(256 * 4);
    float* ldet      = (float*)A(256);

    auto cdiv = [](long a, long b) { return (int)((a + b - 1) / b); };
    const int TPB = 256;

    hipMemsetAsync(ldet, 0, 4, stream);

    // -------- upsample chain: c -> cl[0](1280) -> cl[1](2560) -> cl[2](5120)
    {
        int u[3] = {8, 2, 2}, pw[3] = {11, 2, 2}, Win[3] = {160, 1280, 2560};
        const float* in = C;
        for (int s = 0; s < 3; ++s) {
            long tot = (long)2 * 80 * clT[s];
            k_upsample<<<cdiv(tot, TPB), TPB, 0, stream>>>(in, ups_w[s], ups_b[s], cl[s],
                                                           2, 80, Win[s], u[s], pw[s], clT[s]);
            in = cl[s];
        }
    }

    // -------- squeeze input x -> (2,8,5120)
    k_squeeze_in<<<cdiv(81920, TPB), TPB, 0, stream>>>(X, bufA);

    float* cur = bufA;
    float* nxt = bufB;
    int Bn = 2;
    int dcyc[3] = {6, 5, 4};

    for (int blk = 0; blk < 3; ++blk) {
        int T = 5120 >> blk;
        int g = 1 << blk;
        int ntiles = T >> 4;
        const float* cond = cl[2 - blk];
        // pack cond (2,80,T) -> B frags with K padded to 96
        k_packB<<<cdiv((long)2 * ntiles * 3 * 32, TPB), TPB, 0, stream>>>(
            cond, BfCond, 2, 3, 80, T, 0);

        for (int f = 0; f < 3; ++f) {
            const FlowP& F = fl[blk * 3 + f];
            // actnorm + invertible 1x1
            k_actnorm_inv<<<cdiv((long)Bn * T, TPB), TPB, 0, stream>>>(
                cur, F.an_loc, F.an_scale, F.inv_w, Bn, T);
            k_flow_logdet<<<1, 1, 0, stream>>>(F.an_scale, F.inv_w, ldet, 10240.f);
            // start conv on x_a (rows 0..3)
            k_start<<<cdiv((long)Bn * 128 * T, TPB), TPB, 0, stream>>>(
                cur, F.start_w, F.start_b, audio, Bn, T);
            hipMemsetAsync(skip, 0, (size_t)Bn * 128 * T * 4, stream);

            for (int L = 0; L < 6; ++L) {
                int d = 1 << (L % dcyc[blk]);
                // pack dilated-conv weights: in_w (256,128,3), tap r
                for (int r = 0; r < 3; ++r)
                    k_packA<<<cdiv(16 * 4 * 32, TPB), TPB, 0, stream>>>(
                        F.in_w[L] + r, AfIn + (size_t)r * 32768, 16, 4, 384, 3, 128);
                // pack cond_w layer slice (256 rows, K 80->96)
                k_packA<<<cdiv(16 * 3 * 32, TPB), TPB, 0, stream>>>(
                    F.cond_w + (size_t)L * 256 * 80, AfCond, 16, 3, 80, 1, 80);
                // pack audio into 3 shifted f16 buffers
                for (int r = 0; r < 3; ++r)
                    k_packB<<<cdiv((long)Bn * ntiles * 4 * 32, TPB), TPB, 0, stream>>>(
                        audio, BfAud + (size_t)r * bShiftStride, Bn, 4, 128, T, (r - 1) * d);
                // fused dilated-conv + cond GEMM (60 WMMAs / wave, 16Mx64N)
                long tiles = (long)Bn * 16 * (ntiles >> 2);
                k_gemm_inact<<<cdiv(tiles, 8), TPB, 0, stream>>>(
                    AfIn, BfAud, bShiftStride, AfCond, BfCond,
                    F.in_b[L], F.cond_b, L * 256, F.pos, g, inact, Bn, T);
                // gated tanh*sigmoid
                k_gate<<<cdiv((long)Bn * 128 * T, TPB), TPB, 0, stream>>>(inact, acts, Bn, T);
                // res/skip GEMM
                int last = (L == 5) ? 1 : 0;
                int Mt = last ? 8 : 16;
                k_packB<<<cdiv((long)Bn * ntiles * 4 * 32, TPB), TPB, 0, stream>>>(
                    acts, BfActs, Bn, 4, 128, T, 0);
                k_packA<<<cdiv(Mt * 4 * 32, TPB), TPB, 0, stream>>>(
                    F.rs_w[L], AfRs, Mt, 4, 128, 1, 128);
                tiles = (long)Bn * Mt * (ntiles >> 2);
                k_gemm_rs<<<cdiv(tiles, 8), TPB, 0, stream>>>(
                    AfRs, BfActs, F.rs_b[L], audio, skip, Bn, T, Mt, last);
            }
            // end conv + coupling + log_s reduction
            int nb = cdiv((long)Bn * T, TPB);
            k_end_coupling<<<nb, TPB, 0, stream>>>(skip, F.end_w, F.end_b, cur, partials, Bn, T);
            k_reduce<<<1, 1, 0, stream>>>(partials, nb, ldet);
        }
        if (blk < 2) {
            k_squeeze_batch<<<cdiv(81920, TPB), TPB, 0, stream>>>(cur, nxt, Bn, T);
            float* t2 = cur; cur = nxt; nxt = t2;
            Bn *= 2;
        }
    }

    k_finalize<<<cdiv(81921, TPB), TPB, 0, stream>>>(cur, ldet, (float*)d_out);
}
